// Downsample_Spa_29368986370266
// MI455X (gfx1250) — matrix operand order repaired and checked by reading, not compiled, not run on hardware
//
#include <hip/hip_runtime.h>
#include <hip/hip_bf16.h>
#include <math.h>

typedef __attribute__((ext_vector_type(16))) _Float16 v16h;
typedef __attribute__((ext_vector_type(8)))  _Float16 v8h;
typedef __attribute__((ext_vector_type(8)))  float    v8f;

#define NB   8
#define CH   64
#define H    128
#define W    128
#define OH   64
#define OW   64
#define KK   9
#define KDIM 576     /* CH * KK */
#define KBLK 18      /* KDIM / 32 */

__device__ __forceinline__ int refl(int i) {
    i = (i < 0) ? -i : i;                       // reflect pad=1: -1 -> 1
    return (i > H - 1) ? (2 * (H - 1) - i) : i; // 128 -> 126
}

__global__ __launch_bounds__(256)
void downsample_spa_kernel(const float* __restrict__ x,
                           const float* __restrict__ conv_w,
                           const float* __restrict__ bn_gamma,
                           const float* __restrict__ bn_beta,
                           const float* __restrict__ bn_mean,
                           const float* __restrict__ bn_var,
                           float* __restrict__ out)
{
    // conv weights as f16, rows padded 9->16 with zeros, K permuted to k = off*64 + c
    __shared__ __align__(16) _Float16 wlds[16 * KDIM];
    // raw sigma, then normalized gaussian weights g, per position
    __shared__ float sg[128][KK];

    const int tid = threadIdx.x;
    const int n   = blockIdx.y;        // batch image
    const int oh0 = blockIdx.x * 2;    // this block: output rows oh0, oh0+1

    // ---- stage conv weights into LDS (f16, K-permuted) ----
    for (int i = tid; i < 9 * KDIM; i += 256) {
        int m   = i / KDIM;            // filter index (kk)
        int rem = i - m * KDIM;        // = c*9 + off  (conv_w flat layout)
        int c   = rem / 9;
        int off = rem - c * 9;
        wlds[m * KDIM + off * 64 + c] = (_Float16)conv_w[i];
    }
    for (int i = tid; i < 7 * KDIM; i += 256)   // zero rows 9..15
        wlds[9 * KDIM + i] = (_Float16)0.0f;
    __syncthreads();

    const float* xn = x + (size_t)n * CH * H * W;

    // ---- phase 1: sigma[9 x 128pos] = W[9x576] * patches[576 x pos] via WMMA ----
    {
        const int lane  = tid & 31;
        const int wave  = tid >> 5;        // 8 waves x 16 positions = 128 positions
        const int nn    = lane & 15;
        const int hi    = lane >> 4;
        const int kbase = hi * 16;         // A-fragment K sub-block (contiguous, matches B)
        const int p     = wave * 16 + nn;  // my B column / output position
        const int rowi  = p >> 6;
        const int ow    = p & 63;
        const int oh    = oh0 + rowi;
        const int m     = nn;              // A row (filter index, rows 9..15 zero)

        int rr[3], cc[3];
#pragma unroll
        for (int j = 0; j < 3; ++j) {
            rr[j] = refl(2 * oh - 1 + j);
            cc[j] = refl(2 * ow - 1 + j);
        }

        v8f acc = {};
        for (int kb = 0; kb < KBLK; ++kb) {
            // k = off*64 + c  =>  off constant within a 32-wide k-block
            int off = kb >> 1;
            int kr  = off / 3, kc = off - kr * 3;
            const float* xb = xn + rr[kr] * W + cc[kc];
            int chbase = (kb & 1) * 32;

            // A fragment: lane m holds K = {kbase..kbase+15} (contiguous, kbase = 16*hi)
            const v8h a_lo = *reinterpret_cast<const v8h*>(&wlds[m * KDIM + kb * 32 + kbase]);
            const v8h a_hi = *reinterpret_cast<const v8h*>(&wlds[m * KDIM + kb * 32 + kbase + 8]);
            v16h a, b;
#pragma unroll
            for (int e = 0; e < 8; ++e) { a[e] = a_lo[e]; a[8 + e] = a_hi[e]; }

            // B fragment: lanes 0-15 hold K=0..15, lanes 16-31 hold K=16..31 (col = lane&15)
#pragma unroll
            for (int e = 0; e < 16; ++e) {
                int ch = chbase + hi * 16 + e;       // channel = k & 63
                b[e] = (_Float16)xb[ch * (H * W)];
            }

            acc = __builtin_amdgcn_wmma_f32_16x16x32_f16(
                false, a, false, b, (short)0, acc, false, false);
        }

        // C/D layout: col = lane&15, rows m = vr + 8*hi
        if (hi == 0) {
#pragma unroll
            for (int vr = 0; vr < 8; ++vr) sg[p][vr] = acc[vr];
        } else {
            sg[p][8] = acc[0];
        }
    }
    __syncthreads();

    // ---- BN + clamp + gaussian weights, normalized over the 9 offsets ----
    if (tid < 128) {
        const float d2[9] = {2.f,1.f,2.f,1.f,0.f,1.f,2.f,1.f,2.f};
        float g[9], sum = 0.f;
#pragma unroll
        for (int mm = 0; mm < 9; ++mm) {
            float s = sg[tid][mm];
            s = (s - bn_mean[mm]) * (bn_gamma[mm] * rsqrtf(bn_var[mm] + 1e-5f)) + bn_beta[mm];
            s = fmaxf(s, 1e-4f);
            float q = __expf(-0.5f * d2[mm] / (s * s)) * (0.3989422804014327f / s);
            g[mm] = q; sum += q;
        }
        float inv = 1.f / sum;
#pragma unroll
        for (int mm = 0; mm < 9; ++mm) sg[tid][mm] = g[mm] * inv;
    }
    __syncthreads();

    // ---- phase 2: out[c, oh, ow] = sum_k g[k] * x[c, 2*oh+kr-1, 2*ow+kc-1] ----
    {
        const int pos   = tid & 127;     // position; pairs of threads split channels
        const int chalf = tid >> 7;
        const int rowi  = pos >> 6;
        const int ow    = pos & 63;
        const int oh    = oh0 + rowi;

        float g[9];
#pragma unroll
        for (int mm = 0; mm < 9; ++mm) g[mm] = sg[pos][mm];

        int rr[3], cc[3];
#pragma unroll
        for (int j = 0; j < 3; ++j) {
            rr[j] = refl(2 * oh - 1 + j);
            cc[j] = refl(2 * ow - 1 + j);
        }

        float* outn = out + (size_t)n * CH * OH * OW;
        for (int j = 0; j < 32; ++j) {
            int c = chalf * 32 + j;
            const float* xc = xn + c * (H * W);
            float s = 0.f;
#pragma unroll
            for (int kr = 0; kr < 3; ++kr) {
                const float* xr = xc + rr[kr] * W;
                s = fmaf(g[kr*3+0], xr[cc[0]], s);
                s = fmaf(g[kr*3+1], xr[cc[1]], s);
                s = fmaf(g[kr*3+2], xr[cc[2]], s);
            }
            outn[(c * OH + oh) * OW + ow] = s;
        }
    }
}

extern "C" void kernel_launch(void* const* d_in, const int* in_sizes, int n_in,
                              void* d_out, int out_size, void* d_ws, size_t ws_size,
                              hipStream_t stream) {
    const float* x        = (const float*)d_in[0];
    const float* conv_w   = (const float*)d_in[1];
    const float* bn_gamma = (const float*)d_in[2];
    const float* bn_beta  = (const float*)d_in[3];
    const float* bn_mean  = (const float*)d_in[4];
    const float* bn_var   = (const float*)d_in[5];
    float* out = (float*)d_out;

    dim3 grid(OH / 2, NB);   // 32 oh-pairs x 8 images
    dim3 block(256);
    downsample_spa_kernel<<<grid, block, 0, stream>>>(
        x, conv_w, bn_gamma, bn_beta, bn_mean, bn_var, out);
}
